// ICRewardModel_38654705664107
// MI455X (gfx1250) — compile-verified, matrix operand order checked by reading, not run on hardware
//
#include <hip/hip_runtime.h>
#include <hip/hip_bf16.h>

// ---------------------------------------------------------------------------
// Problem constants (B=2, NL=4, S=512, D=2048, V=32000)
// ---------------------------------------------------------------------------
#define DDIM   2048
#define VOCAB  32000
#define MROWS  4088          // B*NL*(S-1)
#define MPAD   4096          // padded to multiple of 64
#define MBLK   64            // rows per workgroup in the GEMM
#define NTILES 2000          // VOCAB / 16
#define NSLICE 8             // grid.y slices over vocab tiles
#define NT_PER_SLICE (NTILES / NSLICE)   // 250
#define NPART  64            // NSLICE * 8 waves
#define LN_EPS 1e-5f

typedef __attribute__((ext_vector_type(16))) __bf16 v16bf;
typedef __attribute__((ext_vector_type(8)))  __bf16 v8bf;
typedef __attribute__((ext_vector_type(4)))  __bf16 v4bf;
typedef __attribute__((ext_vector_type(8)))  float  v8f;
typedef __attribute__((ext_vector_type(4)))  float  v4f;
typedef unsigned int u32x4 __attribute__((ext_vector_type(4)));
typedef int          i32x4 __attribute__((ext_vector_type(4)));
typedef int          i32x8 __attribute__((ext_vector_type(8)));

// Tensor Data Mover availability (arity differs per toolchain, see CDNA5_HIP.md)
#if defined(__AMDGCN__) && defined(__has_builtin)
#if __has_builtin(__builtin_amdgcn_tensor_load_to_lds) && \
    __has_builtin(__builtin_amdgcn_s_wait_tensorcnt)
#define USE_TDM 1
#endif
#endif

// ---------------------------------------------------------------------------
// Kernel 1: LayerNorm + fp32 -> bf16 staging of hn (rows padded to MPAD)
// ---------------------------------------------------------------------------
__global__ __launch_bounds__(256)
void ln_bf16_kernel(const float* __restrict__ hs, const float* __restrict__ gamma,
                    const float* __restrict__ beta, __bf16* __restrict__ hnb) {
    const int r = blockIdx.x;          // 0..MPAD-1
    const int t = threadIdx.x;         // 0..255
    __shared__ float red[256];

    if (r >= MROWS) {                  // zero-pad dummy rows
        for (int i = t; i < DDIM; i += 256)
            hnb[(size_t)r * DDIM + i] = (__bf16)0.0f;
        return;
    }
    const int b   = r / (4 * 511);
    const int rem = r % (4 * 511);
    const int l   = rem / 511;
    const int s   = rem % 511;
    const float* row = hs + (size_t)((b * 4 + l) * 512 + s) * DDIM;

    float x[8];
    float sum = 0.0f;
    #pragma unroll
    for (int i = 0; i < 8; ++i) { x[i] = row[t + 256 * i]; sum += x[i]; }

    red[t] = sum; __syncthreads();
    for (int o = 128; o > 0; o >>= 1) { if (t < o) red[t] += red[t + o]; __syncthreads(); }
    const float mu = red[0] * (1.0f / DDIM);
    __syncthreads();

    float vs = 0.0f;
    #pragma unroll
    for (int i = 0; i < 8; ++i) { const float d = x[i] - mu; vs += d * d; }
    red[t] = vs; __syncthreads();
    for (int o = 128; o > 0; o >>= 1) { if (t < o) red[t] += red[t + o]; __syncthreads(); }
    const float rinv = rsqrtf(red[0] * (1.0f / DDIM) + LN_EPS);

    #pragma unroll
    for (int i = 0; i < 8; ++i) {
        const int idx = t + 256 * i;
        const float y = (x[i] - mu) * rinv * gamma[idx] + beta[idx];
        hnb[(size_t)r * DDIM + idx] = (__bf16)y;
    }
}

// ---------------------------------------------------------------------------
// Kernel 2: W_head fp32 -> bf16 (65.536M elements); result is L2-resident.
// ---------------------------------------------------------------------------
__global__ __launch_bounds__(256)
void w_to_bf16_kernel(const float* __restrict__ W, __bf16* __restrict__ Wb, long n4) {
    long i = (long)blockIdx.x * blockDim.x + threadIdx.x;
    const long stride = (long)gridDim.x * blockDim.x;
    for (; i < n4; i += stride) {
        const v4f f = ((const v4f*)W)[i];
        v4bf o;
        o[0] = (__bf16)f[0]; o[1] = (__bf16)f[1];
        o[2] = (__bf16)f[2]; o[3] = (__bf16)f[3];
        ((v4bf*)Wb)[i] = o;
    }
}

// ---------------------------------------------------------------------------
// Kernel 3: fused bf16 WMMA GEMM + online logsumexp.
// Grid: (MPAD/MBLK, NSLICE), block 256 (8 waves).
// Each wave owns a distinct set of vocab tiles (stride 8) and multiplies each
// B tile against ALL FOUR 16-row A-slabs (B read exactly once per WG).
// Dynamic LDS: MBLK x DDIM bf16 A tile (256 KB), staged via the TDM.
// ---------------------------------------------------------------------------
__global__ __launch_bounds__(256)
void gemm_lse_kernel(const __bf16* __restrict__ hnb, const __bf16* __restrict__ Wb,
                     float* __restrict__ Mpart, float* __restrict__ Spart) {
    extern __shared__ __bf16 As[];     // [MBLK][DDIM]
    const int mblk = blockIdx.x;
    const int ns   = blockIdx.y;
    const int tid  = threadIdx.x;

#ifdef USE_TDM
    // Stage the 64x2048 bf16 A tile (256 KB) with one Tensor Data Mover op:
    // 1-D tensor of 32768 x 8-byte elements (D# per CDNA5 ISA).
    if (tid < 32) {                    // wave 0 issues the DMA
        const unsigned long long ga =
            (unsigned long long)(size_t)(hnb + (size_t)mblk * MBLK * DDIM);
        u32x4 g0;
        g0[0] = 1u;                                        // count=1, user D#
        g0[1] = (unsigned)(size_t)(&As[0]);                // lds_addr
        g0[2] = (unsigned)(ga & 0xffffffffu);              // global_addr[31:0]
        g0[3] = (unsigned)((ga >> 32) & 0x01ffffffu)       // global_addr[56:32]
              | 0x80000000u;                               // type=2 ("image")
        i32x8 g1 = {};
        g1[0] = 0x00030000;            // data_size=3 (8B), no multicast
        g1[1] = (int)0x80000000;       // tensor_dim0 = 32768 (bits[79:48])
        g1[2] = 0x00010000;            // tensor_dim1 = 1
        g1[3] = (int)0x80000000;       // tile_dim0 = 32768 (bits[127:112])
        g1[4] = 0;                     // tile_dim1/2 unused
        g1[5] = 0x00008000;            // tensor_dim0_stride = 32768
        g1[6] = 0;
        g1[7] = 0;
        i32x4 g2 = {};
        g2[0] = 1;                     // tensor_dim2 = 1
        g2[1] = 1;                     // tensor_dim3 = 1
        g2[2] = 0x00008000;            // tensor_dim2_stride
        i32x4 g3 = {};
        g3[0] = 0x00008000;            // tensor_dim3_stride
        g3[1] = 0x00010000;            // tensor_dim4 = 1
#if defined(__clang_major__) && (__clang_major__ >= 23)
        i32x8 g4 = {};
        __builtin_amdgcn_tensor_load_to_lds(g0, g1, g2, g3, g4, 0);
#else
        __builtin_amdgcn_tensor_load_to_lds(g0, g1, g2, g3, 0);
#endif
        __builtin_amdgcn_s_wait_tensorcnt(0);
    }
#else
    {   // Fallback: cooperative 16-byte copies.
        const v8bf* src = (const v8bf*)(hnb + (size_t)mblk * MBLK * DDIM);
        v8bf* dst = (v8bf*)As;
        for (int i = tid; i < MBLK * DDIM / 8; i += 256) dst[i] = src[i];
    }
#endif
    __syncthreads();

    const int wave  = tid >> 5;        // 0..7 : vocab-tile subset
    const int lane  = tid & 31;
    const int lhalf = lane >> 4;       // 0: K{0..7,16..23}  1: K{8..15,24..31}
    const int l15   = lane & 15;

    const __bf16* A0 = As + (size_t)(0 * 16 + l15) * DDIM;
    const __bf16* A1 = As + (size_t)(1 * 16 + l15) * DDIM;
    const __bf16* A2 = As + (size_t)(2 * 16 + l15) * DDIM;
    const __bf16* A3 = As + (size_t)(3 * 16 + l15) * DDIM;

    float mrun[4][8], srun[4][8];
    #pragma unroll
    for (int q = 0; q < 4; ++q)
        #pragma unroll
        for (int j = 0; j < 8; ++j) { mrun[q][j] = -3.4e38f; srun[q][j] = 0.0f; }

    for (int nt = wave; nt < NT_PER_SLICE; nt += 8) {
        const int n0 = (ns * NT_PER_SLICE + nt) * 16;
        const __bf16* Brow = Wb + (size_t)(n0 + l15) * DDIM;
        if (nt + 8 < NT_PER_SLICE)     // lowers to global_prefetch
            __builtin_prefetch(Wb + (size_t)(n0 + 128 + l15) * DDIM, 0, 0);

        v8f c[4] = {};
        for (int k0 = 0; k0 < DDIM; k0 += 32) {
            const int ka = k0 + lhalf * 8;
            const v8bf blo = *(const v8bf*)(Brow + ka);
            const v8bf bhi = *(const v8bf*)(Brow + ka + 16);
            const v16bf bb = __builtin_shufflevector(blo, bhi,
                0, 1, 2, 3, 4, 5, 6, 7, 8, 9, 10, 11, 12, 13, 14, 15);

            const __bf16* Aq[4] = {A0, A1, A2, A3};
            #pragma unroll
            for (int q = 0; q < 4; ++q) {
                const v8bf alo = *(const v8bf*)(Aq[q] + ka);
                const v8bf ahi = *(const v8bf*)(Aq[q] + ka + 16);
                const v16bf a = __builtin_shufflevector(alo, ahi,
                    0, 1, 2, 3, 4, 5, 6, 7, 8, 9, 10, 11, 12, 13, 14, 15);
                c[q] = __builtin_amdgcn_wmma_f32_16x16x32_bf16(
                           false, a, false, bb, (short)0, c[q], false, false);
            }
        }

        // Online (max, sum-exp) over this tile's 16 vocab columns.
        // C layout: lane holds N=l15; accumulator j holds row q*16 + lhalf*8 + j.
        #pragma unroll
        for (int q = 0; q < 4; ++q) {
            #pragma unroll
            for (int j = 0; j < 8; ++j) {
                const float v = c[q][j];
                float tmax = v;
                for (int msk = 1; msk < 16; msk <<= 1)
                    tmax = fmaxf(tmax, __shfl_xor(tmax, msk, 32));
                const float newm = fmaxf(mrun[q][j], tmax);
                float e = __expf(v - newm);
                for (int msk = 1; msk < 16; msk <<= 1)
                    e += __shfl_xor(e, msk, 32);
                srun[q][j] = srun[q][j] * __expf(mrun[q][j] - newm) + e;
                mrun[q][j] = newm;
            }
        }
    }

    // One lane per 16-lane group writes its 32 row partials for this N subset.
    const int nsEff = ns * 8 + wave;   // 0..NPART-1
    if (l15 == 0) {
        #pragma unroll
        for (int q = 0; q < 4; ++q) {
            const int rbase = mblk * MBLK + q * 16 + lhalf * 8;
            #pragma unroll
            for (int j = 0; j < 8; ++j) {
                Mpart[(size_t)nsEff * MPAD + rbase + j] = mrun[q][j];
                Spart[(size_t)nsEff * MPAD + rbase + j] = srun[q][j];
            }
        }
    }
}

// ---------------------------------------------------------------------------
// Kernel 4: target logit — one wave per row.
// ---------------------------------------------------------------------------
__global__ __launch_bounds__(256)
void target_dot_kernel(const __bf16* __restrict__ hnb, const __bf16* __restrict__ Wb,
                       const int* __restrict__ resp, float* __restrict__ logit_t) {
    const int wave = threadIdx.x >> 5;
    const int lane = threadIdx.x & 31;
    const int r = blockIdx.x * 8 + wave;
    if (r >= MROWS) return;
    const int b   = r / (4 * 511);
    const int s   = (r % (4 * 511)) % 511;
    const int tgt = resp[b * 512 + s + 1];

    const __bf16* a = hnb + (size_t)r * DDIM;
    const __bf16* w = Wb + (size_t)tgt * DDIM;
    float acc = 0.0f;
    for (int k = lane; k < DDIM; k += 32)
        acc += (float)a[k] * (float)w[k];
    for (int msk = 1; msk < 32; msk <<= 1)
        acc += __shfl_xor(acc, msk, 32);
    if (lane == 0) logit_t[r] = acc;
}

// ---------------------------------------------------------------------------
// Kernel 5: combine logsumexp partials -> p = exp(logit_t - logZ)
// ---------------------------------------------------------------------------
__global__ __launch_bounds__(256)
void finalize_p_kernel(const float* __restrict__ Mpart, const float* __restrict__ Spart,
                       const float* __restrict__ logit_t, float* __restrict__ p) {
    const int r = blockIdx.x * 256 + threadIdx.x;
    if (r >= MROWS) return;
    float mg = -3.4e38f;
    #pragma unroll 8
    for (int i = 0; i < NPART; ++i)
        mg = fmaxf(mg, Mpart[(size_t)i * MPAD + r]);
    float Z = 0.0f;
    #pragma unroll 8
    for (int i = 0; i < NPART; ++i)
        Z += Spart[(size_t)i * MPAD + r] * __expf(Mpart[(size_t)i * MPAD + r] - mg);
    p[r] = __expf(logit_t[r] - mg - __logf(Z));
}

// ---------------------------------------------------------------------------
// Kernel 6: reward reduction -> out[b] = mean_s( sum_l p*w_l ) + b_reward
// ---------------------------------------------------------------------------
__global__ __launch_bounds__(256)
void reduce_out_kernel(const float* __restrict__ p, const float* __restrict__ w_reward,
                       const float* __restrict__ b_reward, float* __restrict__ out) {
    const int b = blockIdx.x;
    const int t = threadIdx.x;
    __shared__ float red[256];
    float acc = 0.0f;
    for (int s = t; s < 511; s += 256) {
        float v = 0.0f;
        #pragma unroll
        for (int l = 0; l < 4; ++l)
            v += p[(size_t)((b * 4 + l) * 511) + s] * w_reward[l];
        acc += v;
    }
    red[t] = acc; __syncthreads();
    for (int o = 128; o > 0; o >>= 1) { if (t < o) red[t] += red[t + o]; __syncthreads(); }
    if (t == 0) out[b] = red[0] / 511.0f + b_reward[0];
}

// ---------------------------------------------------------------------------
// Host launcher
// ---------------------------------------------------------------------------
extern "C" void kernel_launch(void* const* d_in, const int* in_sizes, int n_in,
                              void* d_out, int out_size, void* d_ws, size_t ws_size,
                              hipStream_t stream) {
    const float* hs    = (const float*)d_in[0];   // hidden_states
    const float* gamma = (const float*)d_in[1];
    const float* beta  = (const float*)d_in[2];
    const float* Whead = (const float*)d_in[3];
    const float* w_rew = (const float*)d_in[4];
    const float* b_rew = (const float*)d_in[5];
    const int*   resp  = (const int*)d_in[6];
    float* out = (float*)d_out;

    // Workspace layout (bytes), all 16B aligned; total ~150 MB.
    char* ws = (char*)d_ws;
    __bf16* Wb    = (__bf16*)(ws);                          // 131,072,000
    __bf16* hnb   = (__bf16*)(ws + 131072000);              //  16,777,216
    float*  Mpart = (float*)(ws + 147849216);               //   1,048,576
    float*  Spart = (float*)(ws + 148897792);               //   1,048,576
    float*  lt    = (float*)(ws + 149946368);               //      16,384
    float*  pbuf  = (float*)(ws + 149962752);               //      16,384

    // 256 KB dynamic LDS for the A tile (WGP supports 320 KB).
    (void)hipFuncSetAttribute((const void*)gemm_lse_kernel,
                              hipFuncAttributeMaxDynamicSharedMemorySize,
                              MBLK * DDIM * (int)sizeof(__bf16));

    ln_bf16_kernel<<<MPAD, 256, 0, stream>>>(hs, gamma, beta, hnb);
    w_to_bf16_kernel<<<8192, 256, 0, stream>>>(Whead, Wb, (long)VOCAB * DDIM / 4);
    gemm_lse_kernel<<<dim3(MPAD / MBLK, NSLICE), 256,
                      MBLK * DDIM * sizeof(__bf16), stream>>>(hnb, Wb, Mpart, Spart);
    target_dot_kernel<<<(MROWS + 7) / 8, 256, 0, stream>>>(hnb, Wb, resp, lt);
    finalize_p_kernel<<<MPAD / 256, 256, 0, stream>>>(Mpart, Spart, lt, pbuf);
    reduce_out_kernel<<<2, 256, 0, stream>>>(pbuf, w_rew, b_rew, out);
}
